// RNNDecoder_87754771792084
// MI455X (gfx1250) — matrix operand order repaired and checked by reading, not compiled, not run on hardware
//
#include <hip/hip_runtime.h>
#include <hip/hip_bf16.h>

// ---------------------------------------------------------------------------
// Types for CDNA5 WMMA (wave32, 16x16x32 bf16 -> f32) and TDM descriptors
// ---------------------------------------------------------------------------
typedef __bf16 bf16;
typedef bf16 bf16x16 __attribute__((ext_vector_type(16)));
typedef bf16 bf16x8  __attribute__((ext_vector_type(8)));
typedef float floatx8 __attribute__((ext_vector_type(8)));
typedef unsigned int uint32x4 __attribute__((ext_vector_type(4)));
typedef int int32x4 __attribute__((ext_vector_type(4)));
typedef int int32x8 __attribute__((ext_vector_type(8)));

#define E_DIM 512
#define H_DIM 512
#define M_DIM 512
#define V_DIM 20000
#define B_DIM 16
#define T_DIM 128
#define S_DIM 128

__device__ __forceinline__ floatx8 wmma_bf16(bf16x16 a, bf16x16 b, floatx8 c) {
  // (neg_a, A, neg_b, B, c_mod, C, reuse_a, reuse_b)
  return __builtin_amdgcn_wmma_f32_16x16x32_bf16(false, a, false, b, (short)0, c,
                                                 false, false);
}

// A fragment: 16x32 bf16, row = lane&15.
// Lanes 0-15 hold K = k0+0..15; lanes 16-31 hold K = k0+16..31.
__device__ __forceinline__ bf16x16 load_a_frag(const bf16* base, int ldk, int k0, int lane) {
  const bf16* p = base + (size_t)(lane & 15) * ldk + k0 + ((lane & 16) ? 16 : 0);
  return *(const bf16x16*)p;
}

// B fragment: 32x16 bf16 from row-major W[N,K] (k contiguous).
// Lanes 0-15: col n0+lane, K = k0+0..15; lanes 16-31: col n0+lane-16, K = k0+16..31.
__device__ __forceinline__ bf16x16 load_b_frag(const bf16* w, int ldk, int n0, int k0, int lane) {
  const bf16* p = w + (size_t)(n0 + (lane & 15)) * ldk + k0 + ((lane & 16) ? 16 : 0);
  return *(const bf16x16*)p;
}

// ---------------------------------------------------------------------------
// Tensor Data Mover: 2-D tile load (bf16 elements) global -> LDS.
// D# per CDNA5 ISA:
//  group0: [1:0]=count=1, [63:32]=lds_addr, [120:64]=global_addr, [127:126]=type=2
//  group1: [17:16]=data_size(1 => 2B), [79:48]=tensor_dim0, [111:80]=tensor_dim1,
//          [127:112]=tile_dim0, [143:128]=tile_dim1, [207:160]=tensor_dim0_stride
//  groups 2/3 + extra group: zero (2-D tensor), workgroup_mask=0 (not in cluster)
// This toolchain's builtin is the 6-arg clang-23 form:
//  (uint32x4 g0, int32x8 g1, int32x4, int32x4, int32x8, i32 cpol)
// ---------------------------------------------------------------------------
__device__ __forceinline__ void tdm_load_2d_bf16(const void* lds_ptr, const bf16* gptr,
                                                 unsigned tensor_d0, unsigned tensor_d1,
                                                 unsigned tile_d0, unsigned tile_d1,
                                                 unsigned stride0) {
  unsigned lds_addr = (unsigned)(unsigned long long)(size_t)lds_ptr;  // LDS aperture: low 32 bits
  unsigned long long ga = (unsigned long long)(size_t)gptr;
  uint32x4 g0;
  g0[0] = 1u;                                              // count=1
  g0[1] = lds_addr;
  g0[2] = (unsigned)(ga & 0xFFFFFFFFu);
  g0[3] = (unsigned)((ga >> 32) & 0x01FFFFFFu) | (2u << 30);  // type=2 ("image")
  int32x8 g1;
  g1[0] = (int)(1u << 16);                                 // data_size=1 -> 2 bytes
  g1[1] = (int)((tensor_d0 & 0xFFFFu) << 16);
  g1[2] = (int)(((tensor_d0 >> 16) & 0xFFFFu) | ((tensor_d1 & 0xFFFFu) << 16));
  g1[3] = (int)(((tensor_d1 >> 16) & 0xFFFFu) | ((tile_d0 & 0xFFFFu) << 16));
  g1[4] = (int)(tile_d1 & 0xFFFFu);                        // tile_dim2 = 0 (unused)
  g1[5] = (int)stride0;                                    // dim0_stride[31:0]
  g1[6] = 0;                                               // dim0_stride[47:32], dim1_stride lo
  g1[7] = 0;
  int32x4 z4 = {0, 0, 0, 0};
  int32x8 z8 = {0, 0, 0, 0, 0, 0, 0, 0};
  __builtin_amdgcn_tensor_load_to_lds(g0, g1, z4, z4, z8, 0);
}

// ---------------------------------------------------------------------------
// f32 -> bf16 conversion
// ---------------------------------------------------------------------------
__global__ void cvt_f32_bf16_kernel(const float* __restrict__ src,
                                    bf16* __restrict__ dst, int n) {
  int i = blockIdx.x * 256 + threadIdx.x;
  if (i < n) dst[i] = (bf16)src[i];
}

// ---------------------------------------------------------------------------
// Recurrent scan: one workgroup (512 threads = 16 waves) loops over T steps.
// Attention (VALU) + GRU gate GEMMs (WMMA, M=16 = batch).
// ---------------------------------------------------------------------------
__global__ __launch_bounds__(512) void rnn_recurrent_kernel(
    const int* __restrict__ tokens, const int* __restrict__ lengths,
    const int* __restrict__ mem_lengths, const float* __restrict__ hidden0,
    const float* __restrict__ attn_memory, const float* __restrict__ embedding,
    const bf16* __restrict__ w_ih_bf, const bf16* __restrict__ w_hh_bf,
    const float* __restrict__ b_ih, const float* __restrict__ b_hh,
    float* __restrict__ outs) {
  extern __shared__ char smem_raw[];
  float* sh_h  = (float*)smem_raw;                       // 8192 f32
  float* sh_sc = sh_h + B_DIM * H_DIM;                   // 2048 f32
  bf16* sh_xbf = (bf16*)(sh_sc + B_DIM * S_DIM);         // 16384 bf16
  bf16* sh_hbf = sh_xbf + B_DIM * 1024;                  // 8192 bf16
  bf16* sh_r   = sh_hbf + B_DIM * H_DIM;                 // 8192 bf16
  bf16* sh_z   = sh_r + B_DIM * H_DIM;                   // 8192 bf16
  bf16* sh_xn  = sh_z + B_DIM * H_DIM;                   // 8192 bf16
  bf16* sh_hn  = sh_xn + B_DIM * H_DIM;                  // 8192 bf16

  const int tid  = threadIdx.x;
  const int lane = tid & 31;
  const int wave = tid >> 5;  // 0..15

  for (int i = tid; i < B_DIM * H_DIM; i += 512) {
    float v = hidden0[i];
    sh_h[i] = v;
    sh_hbf[i] = (bf16)v;
  }
  __syncthreads();

  for (int t = 0; t < T_DIM; ++t) {
    // (a) scores[b,s] = dot(h[b], attn_memory[b,s,:]) with padding mask
    for (int p = tid; p < B_DIM * S_DIM; p += 512) {
      int b = p >> 7, s = p & 127;
      const float4* mem4 = (const float4*)(attn_memory + (size_t)(b * S_DIM + s) * M_DIM);
      const float4* h4   = (const float4*)(sh_h + b * H_DIM);
      float acc = 0.f;
#pragma unroll 8
      for (int k = 0; k < M_DIM / 4; ++k) {
        float4 a = mem4[k], hh = h4[k];
        acc += a.x * hh.x + a.y * hh.y + a.z * hh.z + a.w * hh.w;
      }
      if (s >= mem_lengths[b]) acc = -1e9f;
      sh_sc[p] = acc;
    }
    __syncthreads();

    // (b) softmax over S: one wave per batch row
    {
      float* sc = sh_sc + wave * S_DIM;
      float mx = -3.4e38f;
      for (int s = lane; s < S_DIM; s += 32) mx = fmaxf(mx, sc[s]);
#pragma unroll
      for (int off = 16; off > 0; off >>= 1) mx = fmaxf(mx, __shfl_xor(mx, off, 32));
      float sum = 0.f;
      for (int s = lane; s < S_DIM; s += 32) {
        float e = __expf(sc[s] - mx);
        sc[s] = e;
        sum += e;
      }
#pragma unroll
      for (int off = 16; off > 0; off >>= 1) sum += __shfl_xor(sum, off, 32);
      float inv = 1.f / sum;
      for (int s = lane; s < S_DIM; s += 32) sc[s] *= inv;
    }
    __syncthreads();

    // (c) context ctx[b,m]; write outs[:, :M]
    for (int p = tid; p < B_DIM * M_DIM; p += 512) {
      int b = p >> 9, m = p & 511;
      const float* aw  = sh_sc + b * S_DIM;
      const float* col = attn_memory + (size_t)b * S_DIM * M_DIM + m;
      float acc = 0.f;
#pragma unroll 8
      for (int s = 0; s < S_DIM; ++s) acc += aw[s] * col[(size_t)s * M_DIM];
      bool valid = t < lengths[b];
      outs[(size_t)(b * T_DIM + t) * (M_DIM + H_DIM) + m] = valid ? acc : 0.f;
      sh_xbf[b * (E_DIM + M_DIM) + E_DIM + m] = (bf16)acc;
    }
    // (d) embedding gather -> x[:, :E]
    for (int p = tid; p < B_DIM * E_DIM; p += 512) {
      int b = p >> 9, e = p & 511;
      int tok = tokens[b * T_DIM + t];
      sh_xbf[b * (E_DIM + M_DIM) + e] = (bf16)embedding[(size_t)tok * E_DIM + e];
    }
    __syncthreads();

    // (e) gate GEMMs via WMMA: G1 = x @ w_ihT (K=1024), G2 = h @ w_hhT (K=512)
    for (int tile = wave; tile < 96; tile += 16) {
      int n0 = tile * 16;
      floatx8 accx = {};
#pragma unroll 4
      for (int kk = 0; kk < 32; ++kk) {
        bf16x16 a  = load_a_frag(sh_xbf, 1024, kk * 32, lane);
        bf16x16 bm = load_b_frag(w_ih_bf, 1024, n0, kk * 32, lane);
        accx = wmma_bf16(a, bm, accx);
      }
      floatx8 acch = {};
#pragma unroll 4
      for (int kk = 0; kk < 16; ++kk) {
        bf16x16 a  = load_a_frag(sh_hbf, 512, kk * 32, lane);
        bf16x16 bm = load_b_frag(w_hh_bf, 512, n0, kk * 32, lane);
        acch = wmma_bf16(a, bm, acch);
      }
      int col   = n0 + (lane & 15);
      int rbase = (lane & 16) ? 8 : 0;
      if (col < 2 * H_DIM) {  // r or z gate: fuse x-path and h-path
        float bsum = b_ih[col] + b_hh[col];
        bf16* dst = (col < H_DIM) ? sh_r : sh_z;
        int c = col & (H_DIM - 1);
#pragma unroll
        for (int j = 0; j < 8; ++j) {
          float v = accx[j] + acch[j] + bsum;
          dst[(rbase + j) * H_DIM + c] = (bf16)(1.f / (1.f + __expf(-v)));
        }
      } else {  // n gate: keep x-path and h-path separate
        int c = col - 2 * H_DIM;
        float bx = b_ih[col], bh = b_hh[col];
#pragma unroll
        for (int j = 0; j < 8; ++j) {
          sh_xn[(rbase + j) * H_DIM + c] = (bf16)(accx[j] + bx);
          sh_hn[(rbase + j) * H_DIM + c] = (bf16)(acch[j] + bh);
        }
      }
    }
    __syncthreads();

    // (f) hidden update + outs[:, M:]
    for (int p = tid; p < B_DIM * H_DIM; p += 512) {
      int b = p >> 9, c = p & 511;
      float r = (float)sh_r[p], z = (float)sh_z[p];
      float n = tanhf((float)sh_xn[p] + r * (float)sh_hn[p]);
      float hnew = (1.f - z) * n + z * sh_h[p];
      bool valid = t < lengths[b];
      outs[(size_t)(b * T_DIM + t) * (M_DIM + H_DIM) + M_DIM + c] = valid ? hnew : 0.f;
      float keep = valid ? hnew : sh_h[p];
      sh_h[p]   = keep;
      sh_hbf[p] = (bf16)keep;
    }
    __syncthreads();
  }
}

// ---------------------------------------------------------------------------
// hid = outs @ w1^T + b1   [2048,1024]x[1024,512] -> bf16 [2048,512]
// ---------------------------------------------------------------------------
__global__ __launch_bounds__(256) void proj1_kernel(
    const bf16* __restrict__ outs_bf, const bf16* __restrict__ w1_bf,
    const float* __restrict__ b1, bf16* __restrict__ hid_bf) {
  int lane = threadIdx.x & 31;
  int wid  = (blockIdx.x * 256 + threadIdx.x) >> 5;  // 0..4095
  int mt = wid >> 5;
  int nt = wid & 31;
  int m0 = mt * 16, n0 = nt * 16;
  floatx8 acc = {};
  const bf16* abase = outs_bf + (size_t)m0 * 1024;
#pragma unroll 4
  for (int kk = 0; kk < 32; ++kk) {
    bf16x16 a  = load_a_frag(abase, 1024, kk * 32, lane);
    bf16x16 bm = load_b_frag(w1_bf, 1024, n0, kk * 32, lane);
    acc = wmma_bf16(a, bm, acc);
  }
  int col   = n0 + (lane & 15);
  int rbase = (lane & 16) ? 8 : 0;
  float bias = b1[col];
#pragma unroll
  for (int j = 0; j < 8; ++j)
    hid_bf[(size_t)(m0 + rbase + j) * 512 + col] = (bf16)(acc[j] + bias);
}

// ---------------------------------------------------------------------------
// logits = hid @ w2^T + b2   [2048,512]x[512,20000] -> f32 (d_out)
// Block tile 128(M) x 160(N), 512 threads (16 waves, 5 tiles each).
// K staged in 64-half chunks via TDM tensor_load_to_lds, double buffered:
//   bufA: 2 x [128][64] bf16 (32KB), bufB: 2 x [160][64] bf16 (40KB) = 72KB LDS
// Grid: 16 x 125 = 2000 blocks.
// ---------------------------------------------------------------------------
__global__ __launch_bounds__(512) void logits_kernel(
    const bf16* __restrict__ hid_bf, const bf16* __restrict__ w2_bf,
    const float* __restrict__ b2, float* __restrict__ logits) {
  extern __shared__ char lsm[];
  bf16* bufA = (bf16*)lsm;                 // 2 * 128*64
  bf16* bufB = bufA + 2 * 128 * 64;        // 2 * 160*64
  const int tid = threadIdx.x, lane = tid & 31, wave = tid >> 5;
  const int mblk = blockIdx.x / 125;       // 0..15
  const int nblk = blockIdx.x - mblk * 125;  // 0..124
  const int m0 = mblk * 128, n0 = nblk * 160;

  floatx8 acc[5] = {};

  // prime chunk 0
  if (wave == 0) {
    tdm_load_2d_bf16(bufA, hid_bf + (size_t)m0 * 512, 512, 2048, 64, 128, 512);
    tdm_load_2d_bf16(bufB, w2_bf + (size_t)n0 * 512, 512, V_DIM, 64, 160, 512);
  }

  for (int c = 0; c < 8; ++c) {
    const int cur = c & 1;
    if (wave == 0) {
      if (c < 7) {
        const int nxt = cur ^ 1;
        tdm_load_2d_bf16(bufA + nxt * 128 * 64,
                         hid_bf + (size_t)m0 * 512 + (c + 1) * 64,
                         512, 2048, 64, 128, 512);
        tdm_load_2d_bf16(bufB + nxt * 160 * 64,
                         w2_bf + (size_t)n0 * 512 + (c + 1) * 64,
                         512, V_DIM, 64, 160, 512);
        __builtin_amdgcn_s_wait_tensorcnt(2);  // chunk c complete
      } else {
        __builtin_amdgcn_s_wait_tensorcnt(0);  // last chunk complete
      }
    }
    __syncthreads();  // chunk c visible to all waves

    const bf16* A  = bufA + cur * 128 * 64;
    const bf16* Bm = bufB + cur * 160 * 64;
#pragma unroll
    for (int i = 0; i < 5; ++i) {
      int tile = wave + 16 * i;  // 0..79
      int mt = tile / 10, nt = tile - mt * 10;
#pragma unroll
      for (int ks = 0; ks < 2; ++ks) {
        bf16x16 a  = load_a_frag(A + mt * 16 * 64, 64, ks * 32, lane);
        bf16x16 bm = load_b_frag(Bm, 64, nt * 16, ks * 32, lane);
        acc[i] = wmma_bf16(a, bm, acc[i]);
      }
    }
    __syncthreads();  // everyone done reading before buffer reuse
  }

  // writeback with bias
  int rbase = (lane & 16) ? 8 : 0;
#pragma unroll
  for (int i = 0; i < 5; ++i) {
    int tile = wave + 16 * i;
    int mt = tile / 10, nt = tile - mt * 10;
    int col = n0 + nt * 16 + (lane & 15);
    float bias = b2[col];
#pragma unroll
    for (int j = 0; j < 8; ++j)
      logits[(size_t)(m0 + mt * 16 + rbase + j) * V_DIM + col] = acc[i][j] + bias;
  }
}

// ---------------------------------------------------------------------------
// In-place log_softmax over V=20000 per row; one block per row.
// ---------------------------------------------------------------------------
__global__ __launch_bounds__(256) void logsoftmax_kernel(float* __restrict__ logits) {
  const int tid = threadIdx.x, lane = tid & 31, wave = tid >> 5;
  float* x = logits + (size_t)blockIdx.x * V_DIM;
  __shared__ float red[8];

  float mx = -3.4e38f;
  for (int i = tid; i < V_DIM; i += 256) mx = fmaxf(mx, x[i]);
#pragma unroll
  for (int off = 16; off > 0; off >>= 1) mx = fmaxf(mx, __shfl_xor(mx, off, 32));
  if (lane == 0) red[wave] = mx;
  __syncthreads();
  mx = red[0];
#pragma unroll
  for (int w = 1; w < 8; ++w) mx = fmaxf(mx, red[w]);
  __syncthreads();

  float sum = 0.f;
  for (int i = tid; i < V_DIM; i += 256) sum += __expf(x[i] - mx);
#pragma unroll
  for (int off = 16; off > 0; off >>= 1) sum += __shfl_xor(sum, off, 32);
  if (lane == 0) red[wave] = sum;
  __syncthreads();
  sum = red[0];
#pragma unroll
  for (int w = 1; w < 8; ++w) sum += red[w];

  float lse = mx + __logf(sum);
  for (int i = tid; i < V_DIM; i += 256) x[i] -= lse;
}

// ---------------------------------------------------------------------------
// Launch
// ---------------------------------------------------------------------------
extern "C" void kernel_launch(void* const* d_in, const int* in_sizes, int n_in,
                              void* d_out, int out_size, void* d_ws, size_t ws_size,
                              hipStream_t stream) {
  const int*   tokens      = (const int*)d_in[0];
  const int*   lengths     = (const int*)d_in[1];
  const int*   mem_lengths = (const int*)d_in[2];
  const float* hidden0     = (const float*)d_in[3];
  const float* attn_memory = (const float*)d_in[4];
  const float* embedding   = (const float*)d_in[5];
  const float* w_ih        = (const float*)d_in[6];
  const float* w_hh        = (const float*)d_in[7];
  const float* b_ih        = (const float*)d_in[8];
  const float* b_hh        = (const float*)d_in[9];
  const float* w1          = (const float*)d_in[10];
  const float* b1          = (const float*)d_in[11];
  const float* w2          = (const float*)d_in[12];
  const float* b2          = (const float*)d_in[13];
  float* out = (float*)d_out;

  char* ws = (char*)d_ws;
  size_t off = 0;
  auto take = [&](size_t bytes) -> char* {
    char* p = ws + off;
    off += (bytes + 255) & ~(size_t)255;
    return p;
  };
  bf16*  w_ih_bf = (bf16*)take((size_t)1536 * 1024 * 2);
  bf16*  w_hh_bf = (bf16*)take((size_t)1536 * 512 * 2);
  bf16*  w1_bf   = (bf16*)take((size_t)512 * 1024 * 2);
  bf16*  w2_bf   = (bf16*)take((size_t)V_DIM * 512 * 2);
  float* outs    = (float*)take((size_t)2048 * 1024 * 4);
  bf16*  outs_bf = (bf16*)take((size_t)2048 * 1024 * 2);
  bf16*  hid_bf  = (bf16*)take((size_t)2048 * 512 * 2);

  auto cvt = [&](const float* s, bf16* d, int n) {
    cvt_f32_bf16_kernel<<<(n + 255) / 256, 256, 0, stream>>>(s, d, n);
  };
  cvt(w_ih, w_ih_bf, 1536 * 1024);
  cvt(w_hh, w_hh_bf, 1536 * 512);
  cvt(w1, w1_bf, 512 * 1024);
  cvt(w2, w2_bf, V_DIM * 512);

  // 152KB dynamic LDS (<160KB)
  rnn_recurrent_kernel<<<1, 512, 155648, stream>>>(
      tokens, lengths, mem_lengths, hidden0, attn_memory, embedding,
      w_ih_bf, w_hh_bf, b_ih, b_hh, outs);

  cvt(outs, outs_bf, 2048 * 1024);

  proj1_kernel<<<512, 256, 0, stream>>>(outs_bf, w1_bf, b1, hid_bf);
  // 72KB dynamic LDS for double-buffered TDM staging
  logits_kernel<<<2000, 512, 73728, stream>>>(hid_bf, w2_bf, b2, out);
  logsoftmax_kernel<<<2048, 256, 0, stream>>>(out);
}
